// CACRFS3D_66400194396211
// MI455X (gfx1250) — matrix operand order repaired and checked by reading, not compiled, not run on hardware
//
#include <hip/hip_runtime.h>
#include <math.h>
#include <stdint.h>

// ---------------- CDNA5 WMMA / TDM types ----------------
typedef __attribute__((ext_vector_type(16))) _Float16 v16h;
typedef __attribute__((ext_vector_type(8)))  float    v8f;
typedef __attribute__((ext_vector_type(2)))  float    v2f;
typedef __attribute__((ext_vector_type(4)))  unsigned int u32x4;
typedef __attribute__((ext_vector_type(8)))  int          i32x8;
typedef __attribute__((ext_vector_type(4)))  int          i32x4;

// ---------------- Problem constants ----------------
#define NWAY   2
#define NPTS   2048
#define FDIM   192
#define NSUB   100
#define KCON   200
#define NQ     2
#define NCLS   3
#define NPROTO 300                 // 3 classes * 100 sub-prototypes
#define NNODE  (NPROTO + NQ*NPTS)  // 4396
#define NPAD   4416                // 276*16 = 138*32, WMMA-friendly padding
#define ALPHA  0.99f
#define NITER  160                 // Chebyshev iterations (kappa<=199 -> conv ~0.867^k)
#define NEGBIG (-1.0e10f)
#define FMAXF  3.402823466e38f

// ---------------- helpers ----------------
__device__ __forceinline__ int block_argmax(float v, int idx, float* rv, int* ri, int tid) {
  rv[tid] = v; ri[tid] = idx;
  __syncthreads();
  for (int s = 128; s > 0; s >>= 1) {
    if (tid < s) {
      float v2 = rv[tid + s]; int i2 = ri[tid + s];
      if (v2 > rv[tid] || (v2 == rv[tid] && i2 < ri[tid])) { rv[tid] = v2; ri[tid] = i2; }
    }
    __syncthreads();
  }
  int r = ri[0];
  __syncthreads();
  return r;
}

// Tensor Data Mover: DMA a 32x32 f32 tile (row stride NPAD elements) from global
// into LDS, inserting 1 DWORD of padding after every 32 DWORDs so the LDS image
// matches a float[32][33] array (bank-conflict-free transposed reads).
// D# layout per CDNA5 ISA ch.8 (group0: count/lds/global/type, group1: dims+strides).
// This toolchain exposes the 6-arg builtin (clang-23 / therock form).
__device__ __forceinline__ void tdm_load_tile32(unsigned lds_byte_off, const float* gptr) {
  unsigned long long ga = (unsigned long long)(uintptr_t)gptr;
  u32x4 g0;
  g0[0] = 1u;                                            // [1:0] count=1 (valid user D#)
  g0[1] = lds_byte_off;                                  // [63:32] lds_addr (bytes)
  g0[2] = (unsigned)(ga & 0xffffffffull);                // [95:64] global_addr lo
  g0[3] = (unsigned)((ga >> 32) & 0x01ffffffull)         // [120:96] global_addr hi
        | (2u << 30);                                    // [127:126] type=2 ("image")
  i32x8 g1;
  g1[0] = (int)((2u << 16)                               // data_size = 2 -> 4 bytes
              | (1u << 20)                               // pad_enable
              | (4u << 22));                             // pad_interval code 4 = 32 DWORDs
                                                         // pad_amount  code 0 = 1 DWORD
  g1[1] = (int)((NPAD & 0xffffu) << 16);                 // tensor_dim0 lo16 @ bits[63:48]
  g1[2] = (int)(((NPAD >> 16) & 0xffffu)                 // tensor_dim0 hi16
              | ((NPAD & 0xffffu) << 16));               // tensor_dim1 lo16
  g1[3] = (int)(((NPAD >> 16) & 0xffffu)                 // tensor_dim1 hi16
              | (32u << 16));                            // tile_dim0 = 32
  g1[4] = (int)(32u);                                    // tile_dim1 = 32, tile_dim2 = 0
  g1[5] = (int)(NPAD);                                   // tensor_dim0_stride lo32 (elements)
  g1[6] = 0;                                             // stride0 hi / stride1 lo
  g1[7] = 0;                                             // stride1 hi
  i32x4 z4 = {0, 0, 0, 0};                               // groups 2/3 unused (2D tensor)
  i32x8 z8 = {0, 0, 0, 0, 0, 0, 0, 0};
  __builtin_amdgcn_tensor_load_to_lds(g0, g1, z4, z4, z8, 0);
}

__global__ void k_zero(float* p, int n) {
  int i = blockIdx.x * blockDim.x + threadIdx.x;
  if (i < n) p[i] = 0.f;
}

// ---------------- stage 0: transpose support features ----------------
// support_feat (NWAY,1,FDIM,NPTS) -> feats (NWAY*NPTS, FDIM) row-major
__global__ void k_transpose_support(const float* __restrict__ sf, float* __restrict__ feats) {
  int idx = blockIdx.x * blockDim.x + threadIdx.x;
  const int total = NWAY * NPTS * FDIM;
  if (idx >= total) return;
  int w   = idx / (NPTS * FDIM);
  int rem = idx % (NPTS * FDIM);
  int p   = rem / FDIM;
  int d   = rem % FDIM;
  feats[idx] = sf[((size_t)w * FDIM + d) * NPTS + p];
}

// ---------------- stage 1: farthest point sampling (1 block per class) ----------------
__global__ void k_fps(const float* __restrict__ feats, const int* __restrict__ fg,
                      int* __restrict__ seeds) {
  const int c   = blockIdx.x;
  const int tid = threadIdx.x;
  const int n    = (c == 0) ? (NWAY * NPTS) : NPTS;
  const int base = (c == 0) ? 0 : (c - 1) * NPTS;

  __shared__ float mind[NWAY * NPTS];
  __shared__ float cur[FDIM];
  __shared__ float rv[256];
  __shared__ int   ri[256];

  // start = argmax(mask) == first masked index (jnp.argmax tie -> first)
  float bv = -1.f; int bi = 0x7fffffff;
  for (int j = tid; j < n; j += 256) {
    bool m = (c == 0) ? (fg[j] == 0) : (fg[base + j] != 0);
    float v = m ? 1.f : 0.f;
    if (v > bv || (v == bv && j < bi)) { bv = v; bi = j; }
  }
  int start = block_argmax(bv, bi, rv, ri, tid);
  if (tid == 0) seeds[c * NSUB + 0] = start;

  for (int d = tid; d < FDIM; d += 256) cur[d] = feats[(size_t)(base + start) * FDIM + d];
  __syncthreads();

  for (int j = tid; j < n; j += 256) {
    const float* x = feats + (size_t)(base + j) * FDIM;
    float s = 0.f;
#pragma unroll 8
    for (int d = 0; d < FDIM; d++) { float t = x[d] - cur[d]; s += t * t; }
    bool m = (c == 0) ? (fg[j] == 0) : (fg[base + j] != 0);
    mind[j] = m ? s : NEGBIG;
  }
  __syncthreads();

  for (int s = 1; s < NSUB; s++) {
    float mv = -FMAXF; int mi = 0x7fffffff;
    for (int j = tid; j < n; j += 256) {
      float v = mind[j];
      if (v > mv || (v == mv && j < mi)) { mv = v; mi = j; }
    }
    int nxt = block_argmax(mv, mi, rv, ri, tid);
    if (tid == 0) seeds[c * NSUB + s] = nxt;
    for (int d = tid; d < FDIM; d += 256) cur[d] = feats[(size_t)(base + nxt) * FDIM + d];
    __syncthreads();
    for (int j = tid; j < n; j += 256) {
      const float* x = feats + (size_t)(base + j) * FDIM;
      float dn = 0.f;
#pragma unroll 8
      for (int d = 0; d < FDIM; d++) { float t = x[d] - cur[d]; dn += t * t; }
      bool m = (c == 0) ? (fg[j] == 0) : (fg[base + j] != 0);
      mind[j] = fminf(mind[j], m ? dn : NEGBIG);
    }
    __syncthreads();
  }
}

// ---------------- stage 2: nearest-seed assignment + scatter mean ----------------
__global__ void k_assign(const float* __restrict__ feats, const int* __restrict__ fg,
                         const int* __restrict__ seeds,
                         float* __restrict__ sums, float* __restrict__ cnt) {
  const int c = blockIdx.y;
  const int n    = (c == 0) ? (NWAY * NPTS) : NPTS;
  const int base = (c == 0) ? 0 : (c - 1) * NPTS;
  int j = blockIdx.x * blockDim.x + threadIdx.x;
  if (j >= n) return;
  bool m = (c == 0) ? (fg[j] == 0) : (fg[base + j] != 0);
  if (!m) return;
  const float* x = feats + (size_t)(base + j) * FDIM;
  float bd = FMAXF; int bs = 0;
  for (int s = 0; s < NSUB; s++) {
    const float* sfp = feats + (size_t)(base + seeds[c * NSUB + s]) * FDIM;
    float d2 = 0.f;
#pragma unroll 8
    for (int d = 0; d < FDIM; d++) { float t = x[d] - sfp[d]; d2 += t * t; }
    if (d2 < bd) { bd = d2; bs = s; }
  }
  int g = c * NSUB + bs;
  for (int d = 0; d < FDIM; d++) atomicAdd(&sums[(size_t)g * FDIM + d], x[d]);
  atomicAdd(&cnt[g], 1.0f);
}

// ---------------- stage 3: assemble node features (f32 + f16) & sqnorms ----------------
__global__ void k_build_nodes(const float* __restrict__ sums, const float* __restrict__ cnt,
                              const float* __restrict__ query,
                              float* __restrict__ node32, _Float16* __restrict__ node16,
                              float* __restrict__ sqn) {
  int r = blockIdx.x;          // 0..NPAD-1
  int d = threadIdx.x;         // 256 threads, first FDIM active
  __shared__ float red[256];
  float v = 0.f;
  if (d < FDIM) {
    if (r < NPROTO)                 v = sums[(size_t)r * FDIM + d] / cnt[r];
    else if (r < NPROTO + NQ*NPTS)  v = query[(size_t)(r - NPROTO) * FDIM + d];
    node32[(size_t)r * FDIM + d] = v;
    node16[(size_t)r * FDIM + d] = (_Float16)v;
  }
  red[d] = v * v;
  __syncthreads();
  for (int s = 128; s > 0; s >>= 1) { if (d < s) red[d] += red[d + s]; __syncthreads(); }
  if (d == 0) sqn[r] = red[0];
}

// ---------------- stage 4: Gram matrix via WMMA f16 -> pairwise d2 ----------------
// One wave per 16x16 output tile. A = X[tm:tm+16, :], B = X^T tile (row segment of X).
__global__ void k_gram(const _Float16* __restrict__ nf, const float* __restrict__ sqn,
                       float* __restrict__ mat) {
  const int tm = blockIdx.x * 16, tn = blockIdx.y * 16;
  const int l  = threadIdx.x;
  const int rA = tm + (l & 15);
  const int rB = tn + (l & 15);
  const int ka = (l < 16) ? 0 : 8;
  const int kb = (l < 16) ? 0 : 8;
  v8f c = {};
  for (int k0 = 0; k0 < FDIM; k0 += 32) {
    const _Float16* ap = nf + (size_t)rA * FDIM + k0 + ka;
    const _Float16* bp = nf + (size_t)rB * FDIM + k0 + kb;
    v16h a, b;
#pragma unroll
    for (int e = 0; e < 8; e++) { a[e] = ap[e]; a[8 + e] = ap[16 + e]; }
#pragma unroll
    for (int e = 0; e < 8; e++) { b[e] = bp[e]; b[8 + e] = bp[16 + e]; }
    c = __builtin_amdgcn_wmma_f32_16x16x32_f16(false, a, false, b, (short)0, c, false, false);
  }
#pragma unroll
  for (int v = 0; v < 8; v++) {
    int row = tm + v + ((l < 16) ? 0 : 8);
    int col = tn + (l & 15);
    float d2 = fmaxf(sqn[row] + sqn[col] - 2.0f * c[v], 0.f);
    if (row >= NNODE || col >= NNODE) d2 = FMAXF;   // padding never selected by kNN
    mat[(size_t)row * NPAD + col] = d2;
  }
}

// ---------------- stage 5: per-row radix-select (k+1 smallest) + sparse affinity ----------------
__global__ void k_topk_a0(float* __restrict__ mat) {
  const int row = blockIdx.x, tid = threadIdx.x;
  float* rp = mat + (size_t)row * NPAD;
  if (row >= NNODE) {                        // padded rows: zero affinity
    for (int j = tid; j < NPAD; j += 256) rp[j] = 0.f;
    return;
  }
  __shared__ unsigned hist[256];
  __shared__ unsigned s_prefix, s_kk;
  if (tid == 0) { s_prefix = 0u; s_kk = KCON + 1; }   // 201st smallest incl. self (d2=0)
  __syncthreads();
  for (int pass = 3; pass >= 0; pass--) {
    hist[tid] = 0u;
    __syncthreads();
    const unsigned prefix = s_prefix;
    const unsigned maskhi = (pass == 3) ? 0u : (0xFFFFFFFFu << ((pass + 1) * 8));
    const int shift = pass * 8;
    for (int j = tid; j < NPAD; j += 256) {
      unsigned u = __float_as_uint(rp[j]);   // all values >= 0 -> bit-monotonic
      if ((u & maskhi) == (prefix & maskhi)) atomicAdd(&hist[(u >> shift) & 255u], 1u);
    }
    __syncthreads();
    if (tid == 0) {
      unsigned kk = s_kk, cum = 0, b = 0;
      for (; b < 256u; b++) { unsigned h = hist[b]; if (cum + h >= kk) break; cum += h; }
      s_prefix = prefix | (b << shift);
      s_kk = kk - cum;
    }
    __syncthreads();
  }
  const unsigned thr = s_prefix;             // exact (k+1)-th smallest key
  for (int j = tid; j < NPAD; j += 256) {
    float d2 = rp[j];
    unsigned u = __float_as_uint(d2);
    rp[j] = (u <= thr && j != row) ? expf(-0.5f * d2) : 0.f;   // self excluded like idx[:,1:]
  }
}

// ---------------- stage 6: degrees of A = A0 + A0^T ----------------
__global__ void k_rowsum(const float* __restrict__ mat, float* __restrict__ rs) {
  const int row = blockIdx.x, tid = threadIdx.x;
  __shared__ float red[256];
  const float* rp = mat + (size_t)row * NPAD;
  float s = 0.f;
  for (int j = tid; j < NPAD; j += 256) s += rp[j];
  red[tid] = s;
  __syncthreads();
  for (int st = 128; st > 0; st >>= 1) { if (tid < st) red[tid] += red[tid + st]; __syncthreads(); }
  if (tid == 0) rs[row] = red[0];
}
__global__ void k_colsum(const float* __restrict__ mat, float* __restrict__ cs) {
  int col = blockIdx.x * blockDim.x + threadIdx.x;
  if (col >= NPAD) return;
  float s = 0.f;
  for (int r = 0; r < NPAD; r++) s += mat[(size_t)r * NPAD + col];
  cs[col] = s;
}
__global__ void k_dis(const float* __restrict__ rs, const float* __restrict__ cs,
                      float* __restrict__ dis) {
  int i = blockIdx.x * blockDim.x + threadIdx.x;
  if (i >= NPAD) return;
  dis[i] = rsqrtf(rs[i] + cs[i] + 1e-8f);
}

// ---------------- stage 7: M = I - alpha * D^-1/2 (A0 + A0^T) D^-1/2, in place ----------------
// Tile-pair (bi<=bj) per block. The two strided 32x32 tiles are fetched by the
// Tensor Data Mover with hardware LDS padding (1 DW per 32 DW) -> float[32][33].
__global__ void k_build_M(float* __restrict__ mat, const float* __restrict__ dis) {
  const int bi = blockIdx.x, bj = blockIdx.y;
  if (bi > bj) return;
  __shared__ float T1[32][33], T2[32][33];
  const int tid = threadIdx.x;
  const int gi0 = bi * 32, gj0 = bj * 32;
  if (tid < 32) {  // wave 0 issues both tile DMAs (TDM ignores EXEC; other waves branch around)
    tdm_load_tile32((unsigned)(uintptr_t)&T1[0][0], mat + (size_t)gi0 * NPAD + gj0);
    tdm_load_tile32((unsigned)(uintptr_t)&T2[0][0], mat + (size_t)gj0 * NPAD + gi0);
    __builtin_amdgcn_s_wait_tensorcnt(0);
  }
  __syncthreads();
  for (int t = tid; t < 1024; t += 256) {
    int r = t >> 5, cc = t & 31;
    int i1 = gi0 + r, j1 = gj0 + cc;
    float a1 = T1[r][cc] + T2[cc][r];
    mat[(size_t)i1 * NPAD + j1] = ((i1 == j1) ? 1.f : 0.f) - ALPHA * dis[i1] * dis[j1] * a1;
    int i2 = gj0 + r, j2 = gi0 + cc;
    float a2 = T2[r][cc] + T1[cc][r];
    mat[(size_t)i2 * NPAD + j2] = ((i2 == j2) ? 1.f : 0.f) - ALPHA * dis[i2] * dis[j2] * a2;
  }
}

// ---------------- stage 8: Chebyshev solve of M Z = Y (3 RHS packed in 16-wide tiles) ----------------
__global__ void k_init_cheb(float* __restrict__ X, float* __restrict__ R, float* __restrict__ Dv) {
  int idx = blockIdx.x * blockDim.x + threadIdx.x;
  if (idx >= NPAD * 16) return;
  int row = idx >> 4, cc = idx & 15;
  float y = (row < NPROTO && cc == (row / NSUB)) ? 1.f : 0.f;  // one-hot labels
  X[idx] = 0.f; R[idx] = y; Dv[idx] = y;                       // x0=0, r0=Y, p0=Y/theta (theta=1)
}
__global__ void k_xupdate(float* __restrict__ X, const float* __restrict__ Dv) {
  int i = blockIdx.x * blockDim.x + threadIdx.x;
  if (i < NPAD * 16) X[i] += Dv[i];
}
__global__ void k_dupdate(float* __restrict__ Dv, const float* __restrict__ R,
                          float c1, float c2) {
  int i = blockIdx.x * blockDim.x + threadIdx.x;
  if (i < NPAD * 16) Dv[i] = c1 * Dv[i] + c2 * R[i];
}
// R -= M * Dv.  4 waves split K; fp32 WMMA 16x16x4; LDS reduce; prefetch the A stream.
__global__ void k_matvec(const float* __restrict__ M, const float* __restrict__ Dv,
                         float* __restrict__ R) {
  __shared__ float part[4][8][32];
  const int wave = threadIdx.x >> 5;
  const int l    = threadIdx.x & 31;
  const int tm   = blockIdx.x * 16;
  const int rA   = tm + (l & 15);
  v8f c = {};
  const int steps = (NPAD / 4) / 4;    // k-blocks per wave
  int kb = wave * steps * 4;
  for (int s = 0; s < steps; s++, kb += 4) {
    v2f a, b;
    const float* ap = M + (size_t)rA * NPAD + kb + ((l < 16) ? 0 : 2);
    __builtin_prefetch(ap + 64, 0, 0);   // pull next 256B of this row's stream toward L0
    a[0] = ap[0]; a[1] = ap[1];
    b[0] = Dv[(kb + ((l < 16) ? 0 : 1)) * 16 + (l & 15)];
    b[1] = Dv[(kb + ((l < 16) ? 2 : 3)) * 16 + (l & 15)];
    c = __builtin_amdgcn_wmma_f32_16x16x4_f32(false, a, false, b, (short)0, c, false, false);
  }
#pragma unroll
  for (int v = 0; v < 8; v++) part[wave][v][l] = c[v];
  __syncthreads();
  if (wave == 0) {
#pragma unroll
    for (int v = 0; v < 8; v++) {
      float s = part[0][v][l] + part[1][v][l] + part[2][v][l] + part[3][v][l];
      int row = tm + v + ((l < 16) ? 0 : 8);
      int col = l & 15;
      R[(size_t)row * 16 + col] -= s;
    }
  }
}

// ---------------- stage 9: pred + loss ----------------
__global__ void k_zero_loss(float* out) {
  if (blockIdx.x == 0 && threadIdx.x == 0) out[NQ * NCLS * NPTS] = 0.f;
}
__global__ void k_out(const float* __restrict__ X, const long long* __restrict__ qy,
                      float* __restrict__ out) {
  int idx = blockIdx.x * blockDim.x + threadIdx.x;   // q*NPTS + p
  __shared__ float red[256];
  float contrib = 0.f;
  if (idx < NQ * NPTS) {
    int q = idx / NPTS, p = idx % NPTS;
    const float* z = X + (size_t)(NPROTO + idx) * 16;
    float z0 = z[0], z1 = z[1], z2 = z[2];
    out[(q * NCLS + 0) * NPTS + p] = z0;             // pred is raw Z transposed
    out[(q * NCLS + 1) * NPTS + p] = z1;
    out[(q * NCLS + 2) * NPTS + p] = z2;
    float mx  = fmaxf(z0, fmaxf(z1, z2));
    float lse = mx + logf(expf(z0 - mx) + expf(z1 - mx) + expf(z2 - mx));
    int   y   = (int)qy[idx];
    float zy  = (y == 0) ? z0 : ((y == 1) ? z1 : z2);
    contrib = -(zy - lse);
  }
  red[threadIdx.x] = contrib;
  __syncthreads();
  for (int s = 128; s > 0; s >>= 1) { if (threadIdx.x < s) red[threadIdx.x] += red[threadIdx.x + s]; __syncthreads(); }
  if (threadIdx.x == 0) atomicAdd(&out[NQ * NCLS * NPTS], red[0] / (float)(NQ * NPTS));
}

// ---------------- host orchestration ----------------
extern "C" void kernel_launch(void* const* d_in, const int* in_sizes, int n_in,
                              void* d_out, int out_size, void* d_ws, size_t ws_size,
                              hipStream_t stream) {
  (void)in_sizes; (void)n_in; (void)out_size; (void)ws_size;
  const float*     sf = (const float*)d_in[0];
  const int*       fg = (const int*)d_in[1];
  const float*     qf = (const float*)d_in[2];
  const long long* qy = (const long long*)d_in[3];
  float* out = (float*)d_out;

  char* ws = (char*)d_ws;
  size_t off = 0;
  auto alloc = [&](size_t bytes) -> char* {
    char* p = ws + off; off += (bytes + 255) & ~(size_t)255; return p;
  };
  float*    mat    = (float*)   alloc((size_t)NPAD * NPAD * 4);    // ~78 MB: fits in 192MB L2
  float*    node32 = (float*)   alloc((size_t)NPAD * FDIM * 4);
  _Float16* node16 = (_Float16*)alloc((size_t)NPAD * FDIM * 2);
  float*    sqn    = (float*)   alloc((size_t)NPAD * 4);
  float*    feats  = (float*)   alloc((size_t)NWAY * NPTS * FDIM * 4);
  int*      seeds  = (int*)     alloc(3 * NSUB * 4);
  float*    sums   = (float*)   alloc((size_t)NPROTO * FDIM * 4);
  float*    cnt    = (float*)   alloc((size_t)NPROTO * 4);
  float*    rs     = (float*)   alloc((size_t)NPAD * 4);
  float*    cs     = (float*)   alloc((size_t)NPAD * 4);
  float*    dis    = (float*)   alloc((size_t)NPAD * 4);
  float*    Xv     = (float*)   alloc((size_t)NPAD * 16 * 4);
  float*    Rv     = (float*)   alloc((size_t)NPAD * 16 * 4);
  float*    Dvv    = (float*)   alloc((size_t)NPAD * 16 * 4);

  k_transpose_support<<<(NWAY * NPTS * FDIM + 255) / 256, 256, 0, stream>>>(sf, feats);
  k_zero<<<(NPROTO * FDIM + 255) / 256, 256, 0, stream>>>(sums, NPROTO * FDIM);
  k_zero<<<(NPROTO + 255) / 256, 256, 0, stream>>>(cnt, NPROTO);
  k_fps<<<3, 256, 0, stream>>>(feats, fg, seeds);
  {
    dim3 ag((NWAY * NPTS + 255) / 256, 3);
    k_assign<<<ag, 256, 0, stream>>>(feats, fg, seeds, sums, cnt);
  }
  k_build_nodes<<<NPAD, 256, 0, stream>>>(sums, cnt, qf, node32, node16, sqn);
  {
    dim3 gg(NPAD / 16, NPAD / 16);
    k_gram<<<gg, 32, 0, stream>>>(node16, sqn, mat);
  }
  k_topk_a0<<<NPAD, 256, 0, stream>>>(mat);
  k_rowsum<<<NPAD, 256, 0, stream>>>(mat, rs);
  k_colsum<<<(NPAD + 255) / 256, 256, 0, stream>>>(mat, cs);
  k_dis<<<(NPAD + 255) / 256, 256, 0, stream>>>(rs, cs, dis);
  {
    dim3 mg(NPAD / 32, NPAD / 32);
    k_build_M<<<mg, 256, 0, stream>>>(mat, dis);
  }
  k_init_cheb<<<(NPAD * 16 + 255) / 256, 256, 0, stream>>>(Xv, Rv, Dvv);

  // Chebyshev semi-iteration: eig(M) in [0.01, 1.99] (normalized adjacency spectrum in [-1,1]).
  const double theta = 1.0, delta = (double)ALPHA;
  const double sigma1 = theta / delta;
  double rho = 1.0 / sigma1;
  for (int it = 0; it < NITER; ++it) {
    k_xupdate<<<(NPAD * 16 + 255) / 256, 256, 0, stream>>>(Xv, Dvv);
    k_matvec<<<NPAD / 16, 128, 0, stream>>>(mat, Dvv, Rv);
    double rho1 = 1.0 / (2.0 * sigma1 - rho);
    float c1 = (float)(rho1 * rho);
    float c2 = (float)(2.0 * rho1 / delta);
    k_dupdate<<<(NPAD * 16 + 255) / 256, 256, 0, stream>>>(Dvv, Rv, c1, c2);
    rho = rho1;
  }

  k_zero_loss<<<1, 32, 0, stream>>>(out);
  k_out<<<(NQ * NPTS + 255) / 256, 256, 0, stream>>>(Xv, qy, out);
}
